// RadianceNet_67585605370023
// MI455X (gfx1250) — matrix operand order repaired and checked by reading, not compiled, not run on hardware
//
#include <hip/hip_runtime.h>
#include <cmath>

typedef __attribute__((ext_vector_type(16))) _Float16 v16h;
typedef __attribute__((ext_vector_type(8)))  _Float16 v8h;
typedef __attribute__((ext_vector_type(8)))  float    v8f;

// ---------------- weight / bias arena layout (element offsets) ----------------
enum : size_t {
  OFF_L0 = 0,                      // 64->256   (Kp=64 , Fp=256)
  OFF_L1 = OFF_L0 + 256 * 64,      // 256->256
  OFF_L2 = OFF_L1 + 256 * 256,
  OFF_L3 = OFF_L2 + 256 * 256,
  OFF_L4 = OFF_L3 + 256 * 256,
  OFF_SG = OFF_L4 + 256 * 256,     // 256->1    (Fp=16)
  OFF_C0 = OFF_SG + 16 * 256,      // 280->128  (Kp=288)
  OFF_C1 = OFF_C0 + 128 * 288,     // 128->128
  OFF_C2 = OFF_C1 + 128 * 128,
  OFF_C3 = OFF_C2 + 128 * 128,     // 128->3    (Fp=16)
  OFF_S0 = OFF_C3 + 16 * 128,      // 256->256
  OFF_S1 = OFF_S0 + 256 * 256,
  OFF_S2 = OFF_S1 + 256 * 256,
  OFF_S3 = OFF_S2 + 256 * 256,     // 256->16
  TOTAL_W = OFF_S3 + 16 * 256      // halfs
};
enum : size_t {
  BO_L0 = 0, BO_L1 = 256, BO_L2 = 512, BO_L3 = 768, BO_L4 = 1024,
  BO_SG = 1280,
  BO_C0 = 1296, BO_C1 = 1424, BO_C2 = 1552, BO_C3 = 1680,
  BO_S0 = 1696, BO_S1 = 1952, BO_S2 = 2208, BO_S3 = 2464,
  TOTAL_B = 2480                   // floats
};

#define LDH 256
#define LDC 128
#define LDPE 32
enum { WAVE_SMEM = 16 * LDH + 16 * LDC + 16 * LDPE };  // 6656 halfs / wave

enum { EPI_LRELU = 0, EPI_SIGMA = 1, EPI_COLOR = 2, EPI_SEM = 3 };

// ---------------- weight repack: W[K][F] f32 -> Wt[Fp][Kp] f16 (zero padded) --
__global__ void pack_weights(const float* __restrict__ W, const float* __restrict__ b,
                             _Float16* __restrict__ Wt, float* __restrict__ bt,
                             int K, int F, int Kp, int Fp) {
  int idx = blockIdx.x * blockDim.x + threadIdx.x;
  int total = Fp * Kp;
  if (idx < total) {
    int f = idx / Kp, k = idx % Kp;
    float v = (f < F && k < K) ? W[(size_t)k * F + f] : 0.0f;
    Wt[idx] = (_Float16)v;
  }
  if (idx < Fp) bt[idx] = (idx < F) ? b[idx] : 0.0f;
}

// A fragment (16x32 f16, CDNA5 layout): lane l holds row m=l&15;
// lanes 0-15: K = kk+0..7 and kk+16..23 ; lanes 16-31: K = kk+8..15 and kk+24..31
__device__ __forceinline__ v16h load_a_frag(const _Float16* base, int ld, int m, int hi, int kk) {
  union { v16h v; v8h h[2]; } a;
  a.h[0] = *(const v8h*)(base + m * ld + kk + hi * 8);
  a.h[1] = *(const v8h*)(base + m * ld + kk + 16 + hi * 8);
  return a.v;
}

// One 16-row GEMM tile across a full layer: D[16,FP] = act(A[16,KP] * Wt^T + b)
// Wt is f16 [FP][KW] row-major, KW = KP (+32 when a PE k-chunk is appended).
template<int KP, int FP, int EPI, bool PE>
__device__ __forceinline__ void wave_gemm(const _Float16* __restrict__ Wt,
                                          const float* __restrict__ Bs,
                                          const _Float16* in, int ldIn,
                                          const _Float16* pe,
                                          _Float16* outLds, int ldOut,
                                          float* __restrict__ gout,
                                          int row0, int lane) {
  constexpr int NK = KP / 32;
  constexpr int NF = FP / 16;
  constexpr int KW = KP + (PE ? 32 : 0);
  const int ln = lane & 15;   // A row index == B/C column index
  const int hi = lane >> 4;

  v8f acc[NF];
  const v8f zero = {0.f, 0.f, 0.f, 0.f, 0.f, 0.f, 0.f, 0.f};
#pragma unroll
  for (int f = 0; f < NF; ++f) acc[f] = zero;

  for (int k = 0; k < NK; ++k) {
    const int kk = k * 32;
    v16h a = load_a_frag(in, ldIn, ln, hi, kk);
#pragma unroll
    for (int f = 0; f < NF; ++f) {
      // B fragment (32x16 f16): lane = column f*16+ln; lanes 0-15 K=kk+0..7,kk+16..23,
      // lanes 16-31 K=kk+8..15,kk+24..31 (same order as A fragment).
      v16h b = load_a_frag(Wt + (size_t)(f * 16) * KW, KW, ln, hi, kk);
      acc[f] = __builtin_amdgcn_wmma_f32_16x16x32_f16(
          false, a, false, b, (short)0, acc[f], false, false);
    }
  }
  if constexpr (PE) {  // extra 32-wide k-chunk from the positional-encoding buffer
    v16h a = load_a_frag(pe, LDPE, ln, hi, 0);
#pragma unroll
    for (int f = 0; f < NF; ++f) {
      v16h b = load_a_frag(Wt + (size_t)(f * 16) * KW, KW, ln, hi, KP);
      acc[f] = __builtin_amdgcn_wmma_f32_16x16x32_f16(
          false, a, false, b, (short)0, acc[f], false, false);
    }
  }

  // C/D layout: VGPR r, lanes 0-15 -> (M=r, N=lane); lanes 16-31 -> (M=8+r, N=lane-16)
  const size_t orow = (size_t)row0 + hi * 8;
  if constexpr (EPI == EPI_LRELU) {
#pragma unroll
    for (int f = 0; f < NF; ++f) {
      float bb = Bs[f * 16 + ln];
#pragma unroll
      for (int r = 0; r < 8; ++r) {
        float x = acc[f][r] + bb;
        x = x > 0.0f ? x : 0.01f * x;
        outLds[(hi * 8 + r) * ldOut + f * 16 + ln] = (_Float16)x;
      }
    }
  } else if constexpr (EPI == EPI_SIGMA) {
    if (ln == 0) {
      float bb = Bs[0];
#pragma unroll
      for (int r = 0; r < 8; ++r) {
        float x = acc[0][r] + bb - 1.0f;                    // softplus(sigma - 1)
        float sp = x > 20.0f ? x : logf(1.0f + expf(x));
        gout[(orow + r) * 20 + 0] = sp;
      }
    }
  } else if constexpr (EPI == EPI_COLOR) {
    if (ln < 3) {
      float bb = Bs[ln];
#pragma unroll
      for (int r = 0; r < 8; ++r) {
        float x = acc[0][r] + bb;
        float s = 1.0f / (1.0f + expf(-x));
        gout[(orow + r) * 20 + 1 + ln] = s * 1.002f - 0.001f;
      }
    }
  } else {  // EPI_SEM: final linear layer, 16 outputs
    float bb = Bs[ln];
#pragma unroll
    for (int r = 0; r < 8; ++r)
      gout[(orow + r) * 20 + 4 + ln] = acc[0][r] + bb;
  }
}

// ---------------- fused radiance network: 4 waves/block, 16 points/wave ------
__global__ __launch_bounds__(128) void radiance_fused(
    const float* __restrict__ feat, const float* __restrict__ vdir,
    const _Float16* __restrict__ W, const float* __restrict__ B,
    float* __restrict__ out, int npts) {
  __shared__ _Float16 smem[4 * WAVE_SMEM];
  const int lane = threadIdx.x & 31;
  const int wave = threadIdx.x >> 5;
  const int row0 = ((int)blockIdx.x * 4 + wave) * 16;
  if (row0 >= npts) return;

  _Float16* bufH = smem + wave * WAVE_SMEM;  // 16 x 256 hidden activations (f16)
  _Float16* bufC = bufH + 16 * LDH;          // 16 x 128 color activations
  _Float16* peB  = bufC + 16 * LDC;          // 16 x 32  positional encoding

  // stage feat tile (16 x 64) into LDS as f16
  for (int j = lane; j < 16 * 64; j += 32) {
    int r = j >> 6, c = j & 63;
    bufH[r * LDH + c] = (_Float16)feat[(size_t)(row0 + r) * 64 + c];
  }
  __syncthreads();

  // block1: 64->256, then 4 x 256->256 (LeakyReLU)
  wave_gemm< 64, 256, EPI_LRELU, false>(W + OFF_L0, B + BO_L0, bufH, LDH, nullptr, bufH, LDH, out, row0, lane);
  __syncthreads();
  wave_gemm<256, 256, EPI_LRELU, false>(W + OFF_L1, B + BO_L1, bufH, LDH, nullptr, bufH, LDH, out, row0, lane);
  __syncthreads();
  wave_gemm<256, 256, EPI_LRELU, false>(W + OFF_L2, B + BO_L2, bufH, LDH, nullptr, bufH, LDH, out, row0, lane);
  __syncthreads();
  wave_gemm<256, 256, EPI_LRELU, false>(W + OFF_L3, B + BO_L3, bufH, LDH, nullptr, bufH, LDH, out, row0, lane);
  __syncthreads();
  wave_gemm<256, 256, EPI_LRELU, false>(W + OFF_L4, B + BO_L4, bufH, LDH, nullptr, bufH, LDH, out, row0, lane);
  __syncthreads();

  // density = softplus(h @ sigma_W + b - 1)
  wave_gemm<256, 16, EPI_SIGMA, false>(W + OFF_SG, B + BO_SG, bufH, LDH, nullptr, nullptr, 0, out, row0, lane);

  // positional encoding of viewdir -> 16 x 24 (cols 24..31 zero; matching W pad)
  for (int j = lane; j < 16 * LDPE; j += 32) peB[j] = (_Float16)0.0f;
  for (int j = lane; j < 16 * 24; j += 32) {
    int r = j / 24, c = j % 24;
    int e = c >> 1, dim = e >> 2, fq = e & 3;   // pts order: dim-major, freq minor
    float p = vdir[(size_t)(row0 + r) * 3 + dim] * (float)(1 << fq);
    peB[r * LDPE + c] = (_Float16)((c & 1) ? cosf(p) : sinf(p));
  }
  __syncthreads();

  // color head: [h | pe] (280 -> pad 288) -> 128 -> 128 -> 128 -> 3
  wave_gemm<256, 128, EPI_LRELU, true >(W + OFF_C0, B + BO_C0, bufH, LDH, peB, bufC, LDC, out, row0, lane);
  __syncthreads();
  wave_gemm<128, 128, EPI_LRELU, false>(W + OFF_C1, B + BO_C1, bufC, LDC, nullptr, bufC, LDC, out, row0, lane);
  __syncthreads();
  wave_gemm<128, 128, EPI_LRELU, false>(W + OFF_C2, B + BO_C2, bufC, LDC, nullptr, bufC, LDC, out, row0, lane);
  __syncthreads();
  wave_gemm<128, 16, EPI_COLOR, false>(W + OFF_C3, B + BO_C3, bufC, LDC, nullptr, nullptr, 0, out, row0, lane);

  // semantic head: 3 x 256->256 (LReLU) then 256->16 linear (h no longer needed)
  wave_gemm<256, 256, EPI_LRELU, false>(W + OFF_S0, B + BO_S0, bufH, LDH, nullptr, bufH, LDH, out, row0, lane);
  __syncthreads();
  wave_gemm<256, 256, EPI_LRELU, false>(W + OFF_S1, B + BO_S1, bufH, LDH, nullptr, bufH, LDH, out, row0, lane);
  __syncthreads();
  wave_gemm<256, 256, EPI_LRELU, false>(W + OFF_S2, B + BO_S2, bufH, LDH, nullptr, bufH, LDH, out, row0, lane);
  __syncthreads();
  wave_gemm<256, 16, EPI_SEM, false>(W + OFF_S3, B + BO_S3, bufH, LDH, nullptr, nullptr, 0, out, row0, lane);
}

extern "C" void kernel_launch(void* const* d_in, const int* in_sizes, int n_in,
                              void* d_out, int out_size, void* d_ws, size_t ws_size,
                              hipStream_t stream) {
  (void)n_in; (void)out_size; (void)ws_size;
  const float* feat = (const float*)d_in[0];
  const float* vdir = (const float*)d_in[1];

  _Float16* wArena = (_Float16*)d_ws;
  float*    bArena = (float*)((char*)d_ws + (size_t)TOTAL_W * sizeof(_Float16));

  // d_in layout: 0 feat, 1 viewdir, 2..11 block1 (W,b)x5, 12/13 sigma,
  //              14..21 color (W,b)x4, 22..29 semantic (W,b)x4
  struct D { int wi; int K, F, Kp, Fp; size_t wo, bo; };
  static const D ds[14] = {
    { 2,  64, 256,  64, 256, OFF_L0, BO_L0},
    { 4, 256, 256, 256, 256, OFF_L1, BO_L1},
    { 6, 256, 256, 256, 256, OFF_L2, BO_L2},
    { 8, 256, 256, 256, 256, OFF_L3, BO_L3},
    {10, 256, 256, 256, 256, OFF_L4, BO_L4},
    {12, 256,   1, 256,  16, OFF_SG, BO_SG},
    {14, 280, 128, 288, 128, OFF_C0, BO_C0},
    {16, 128, 128, 128, 128, OFF_C1, BO_C1},
    {18, 128, 128, 128, 128, OFF_C2, BO_C2},
    {20, 128,   3, 128,  16, OFF_C3, BO_C3},
    {22, 256, 256, 256, 256, OFF_S0, BO_S0},
    {24, 256, 256, 256, 256, OFF_S1, BO_S1},
    {26, 256, 256, 256, 256, OFF_S2, BO_S2},
    {28, 256,  16, 256,  16, OFF_S3, BO_S3},
  };
  for (int i = 0; i < 14; ++i) {
    const D& d = ds[i];
    int total  = d.Fp * d.Kp;
    int blocks = (total + 255) / 256;
    pack_weights<<<blocks, 256, 0, stream>>>(
        (const float*)d_in[d.wi], (const float*)d_in[d.wi + 1],
        wArena + d.wo, bArena + d.bo, d.K, d.F, d.Kp, d.Fp);
  }

  int npts   = in_sizes[0] / 64;          // 2048*128 points
  int blocks = (npts + 63) / 64;          // 64 points / block (4 waves x 16)
  radiance_fused<<<blocks, 128, 0, stream>>>(feat, vdir, wArena, bArena,
                                             (float*)d_out, npts);
}
